// TwistForceField_81046032875769
// MI455X (gfx1250) — matrix operand order repaired and checked by reading, not compiled, not run on hardware
//
#include <hip/hip_runtime.h>
#include <stdint.h>

#define B_  16
#define L_  128
#define R_  512
#define F_  256
#define H_  8
#define HF_ 256
#define FE_ 64
#define E_  16384

typedef __attribute__((ext_vector_type(16))) __bf16          v16bf;
typedef __attribute__((ext_vector_type(8)))  float           v8f;
typedef __attribute__((ext_vector_type(8)))  unsigned short  v8us;

union FragBF { v16bf v; v8us h[2]; };

__device__ __forceinline__ unsigned short f2bfu(float f) {
  union { float f; unsigned u; } x; x.f = f;
  return (unsigned short)((x.u + 0x7FFFu + ((x.u >> 16) & 1u)) >> 16);  // RNE truncate
}
__device__ __forceinline__ float lrelu(float x) { return x > 0.f ? x : 0.01f * x; }

__device__ __forceinline__ v8f wmma_bf16(v16bf a, v16bf b, v8f c) {
  // (neg_a, A, neg_b, B, c_mod, C, reuse_a, reuse_b)
  return __builtin_amdgcn_wmma_f32_16x16x32_bf16(false, a, false, b, (short)0, c, false, false);
}

// A fragment 16x32 bf16 (ISA 7.12.2): lane holds row (lane&15);
// lo 8 elems at k0 + half*8, hi 8 elems at k0 + 16 + half*8.
__device__ __forceinline__ v16bf frag_a(const unsigned short* S, int row0, int stride, int k0) {
  const int l = threadIdx.x & 31;
  const unsigned short* p = S + (size_t)(row0 + (l & 15)) * stride + k0 + ((l >> 4) << 3);
  FragBF f; f.h[0] = *(const v8us*)p; f.h[1] = *(const v8us*)(p + 16);
  return f.v;
}
// B fragment 32x16 bf16: B(k,n) = S[n][k]; lane holds col (lane&15),
// lo 8 elems at k0 + half*8, hi 8 elems at k0 + 16 + half*8.
__device__ __forceinline__ v16bf frag_b(const unsigned short* S, int col0, int stride, int k0) {
  const int l = threadIdx.x & 31;
  const unsigned short* p = S + (size_t)(col0 + (l & 15)) * stride + k0 + ((l >> 4) << 3);
  FragBF f; f.h[0] = *(const v8us*)p; f.h[1] = *(const v8us*)(p + 16);
  return f.v;
}

// CDNA5 async global->LDS DMA copy (16B per lane), ASYNCcnt-tracked.
__device__ __forceinline__ void async_ld16(void* lds_dst, const void* gsrc) {
  unsigned lds = (unsigned)(size_t)lds_dst;
  unsigned long long ga = (unsigned long long)(size_t)gsrc;
  asm volatile("global_load_async_to_lds_b128 %0, %1, off"
               :: "v"(lds), "v"(ga) : "memory");
}
__device__ __forceinline__ void async_wait() {
  asm volatile("s_wait_asynccnt 0x0" ::: "memory");
}

// ---------------------------------------------------------------------------
// Prologue: f32 -> bf16 weight conversion (run once, result reused by all WGs)
// ---------------------------------------------------------------------------
__global__ __launch_bounds__(256) void k_cvt_bf16(
    const float* __restrict__ src, unsigned short* __restrict__ dst, int n)
{
  int i = (blockIdx.x * 256 + threadIdx.x) * 4;
  if (i < n) {
    float4 v = *(const float4*)(src + i);
    unsigned short* d = dst + i;
    d[0] = f2bfu(v.x); d[1] = f2bfu(v.y); d[2] = f2bfu(v.z); d[3] = f2bfu(v.w);
  }
}

// ---------------------------------------------------------------------------
// Kernel 1: AttentionContract. One WG per (b,l).
// M=128 (y), N=256 Wc tiles + one padded Wca tile, K=256 chunked by 64.
// ---------------------------------------------------------------------------
__global__ __launch_bounds__(256) void k_contract(
    const float* __restrict__ ll, const unsigned char* __restrict__ maskp,
    const unsigned short* __restrict__ Wcb,   // bf16 (256,256)
    const unsigned short* __restrict__ Wcab,  // bf16 (16,256), rows 8..15 zero
    const float* __restrict__ bc, const float* __restrict__ gc,
    const float* __restrict__ bec, const float* __restrict__ bca,
    float* __restrict__ outC)
{
  __shared__ unsigned short Xs[128 * 72];   // lrelu(ll_feat) chunk, bf16
  __shared__ unsigned short Ws[256 * 72];   // Wc chunk (reused as f32 partials later)
  __shared__ unsigned short WaS[16 * 72];   // Wca chunk
  __shared__ float atnS[128 * 8];

  const int tid  = threadIdx.x;
  const int lane = tid & 31, wv = tid >> 5;
  const int r15  = lane & 15, hx = lane >> 4;
  const int bl   = blockIdx.x;                       // b*L + l
  const float* llrow = ll + (size_t)bl * (L_ * F_);  // [y][f]

  v8f acc[16] = {};
  v8f accA = {};

  for (int kc = 0; kc < 4; ++kc) {
    __syncthreads();
    // async DMA the bf16 weight chunks straight into LDS
    for (int i = tid; i < 2048; i += 256) {           // 256 rows x 64 cols
      int n = i >> 3, c = (i & 7) << 3;
      async_ld16(&Ws[n * 72 + c], Wcb + (size_t)n * F_ + kc * 64 + c);
    }
    if (tid < 128) {                                   // 16 rows x 64 cols
      int n = tid >> 3, c = (tid & 7) << 3;
      async_ld16(&WaS[n * 72 + c], Wcab + (size_t)n * F_ + kc * 64 + c);
    }
    // stage lrelu(X) chunk: 128 rows x 64 cols, f32 -> bf16
    for (int i = tid; i < 128 * 16; i += 256) {
      int y = i >> 4, c4 = (i & 15) << 2;
      float4 v = *(const float4*)(llrow + y * F_ + kc * 64 + c4);
      unsigned short* d = &Xs[y * 72 + c4];
      d[0] = f2bfu(lrelu(v.x)); d[1] = f2bfu(lrelu(v.y));
      d[2] = f2bfu(lrelu(v.z)); d[3] = f2bfu(lrelu(v.w));
    }
    // prefetch next X chunk (the only real HBM stream)
    if (kc < 3)
      __builtin_prefetch(llrow + (tid >> 1) * F_ + (kc + 1) * 64 + (tid & 1) * 32, 0, 1);
    async_wait();
    __syncthreads();
    #pragma unroll
    for (int ks = 0; ks < 2; ++ks) {
      const int k0 = ks * 32;
      v16bf a  = frag_a(Xs, wv * 16, 72, k0);
      // ping-pong B buffers: keep one fragment load in flight across each WMMA
      v16bf b0 = frag_b(Ws,  0, 72, k0);
      v16bf b1 = frag_b(Ws, 16, 72, k0);
      #pragma unroll
      for (int t = 0; t < 16; ++t) {
        acc[t] = wmma_bf16(a, (t & 1) ? b1 : b0, acc[t]);
        int u = t + 2;
        if (u < 16)       { if (t & 1) b1 = frag_b(Ws, u * 16, 72, k0); else b0 = frag_b(Ws, u * 16, 72, k0); }
        else if (u == 16) { if (t & 1) b1 = frag_b(WaS, 0, 72, k0);     else b0 = frag_b(WaS, 0, 72, k0); }
      }
      accA = wmma_bf16(a, b0, accA);   // tile index 16 (even) lives in b0
    }
  }

  // bias + layernorm; stats via 16-lane shfl_xor reductions
  float s[8] = {}, q[8] = {};
  #pragma unroll
  for (int t = 0; t < 16; ++t) {
    float bn = bc[t * 16 + r15];
    #pragma unroll
    for (int r = 0; r < 8; ++r) {
      float v = acc[t][r] + bn; acc[t][r] = v;
      s[r] += v; q[r] += v * v;
    }
  }
  #pragma unroll
  for (int r = 0; r < 8; ++r) {
    float sv = s[r], qv = q[r];
    #pragma unroll
    for (int m = 1; m < 16; m <<= 1) { sv += __shfl_xor(sv, m, 32); qv += __shfl_xor(qv, m, 32); }
    float mu = sv * (1.f / 256.f);
    s[r] = mu;
    q[r] = rsqrtf(qv * (1.f / 256.f) - mu * mu + 1e-5f);
  }
  #pragma unroll
  for (int t = 0; t < 16; ++t) {
    int n = t * 16 + r15; float g = gc[n], bb = bec[n];
    #pragma unroll
    for (int r = 0; r < 8; ++r) acc[t][r] = g * (acc[t][r] - s[r]) * q[r] + bb;
  }

  // raw attention logits -> LDS (head = lane&15 < 8)
  if (r15 < 8) {
    float bca_h = bca[r15];
    #pragma unroll
    for (int r = 0; r < 8; ++r) {
      int y = wv * 16 + r + 8 * hx;
      atnS[y * 8 + r15] = accA[r] + bca_h;
    }
  }
  __syncthreads();

  // masked softmax over y, one wave per head
  {
    int h = wv;
    float v0[4]; float mx = -3.4e38f;
    #pragma unroll
    for (int j = 0; j < 4; ++j) { v0[j] = atnS[(lane + 32 * j) * 8 + h]; mx = fmaxf(mx, v0[j]); }
    #pragma unroll
    for (int m = 1; m < 32; m <<= 1) mx = fmaxf(mx, __shfl_xor(mx, m, 32));
    float sum = 0.f; float ev[4];
    #pragma unroll
    for (int j = 0; j < 4; ++j) {
      float msk = maskp[(size_t)bl * L_ + lane + 32 * j] ? 1.f : 0.f;
      float e = __expf(v0[j] - mx) * msk;
      ev[j] = e; sum += e;
    }
    #pragma unroll
    for (int m = 1; m < 32; m <<= 1) sum += __shfl_xor(sum, m, 32);
    float inv = 1.f / (sum + 1e-10f);
    #pragma unroll
    for (int j = 0; j < 4; ++j) atnS[(lane + 32 * j) * 8 + h] = ev[j] * inv;
  }
  __syncthreads();

  // weighted sum over y; per-wave partials into (reused) Ws region, tree reduce
  float* cpart = (float*)Ws;
  #pragma unroll
  for (int t = 0; t < 16; ++t) {
    float cp = 0.f; int h = t >> 1;
    #pragma unroll
    for (int r = 0; r < 8; ++r)
      cp += atnS[(wv * 16 + r + 8 * hx) * 8 + h] * acc[t][r];
    cp += __shfl_xor(cp, 16, 32);
    if (hx == 0) cpart[wv * 256 + t * 16 + r15] = cp;
  }
  __syncthreads();
  float o = 0.f;
  #pragma unroll
  for (int w = 0; w < 8; ++w) o += cpart[w * 256 + tid];
  outC[(size_t)bl * HF_ + tid] = o;
}

// ---------------------------------------------------------------------------
// Kernel 2a: NormAndLinear (xf / yf). 128 rows per WG, bf16 out to d_ws.
// ---------------------------------------------------------------------------
__global__ __launch_bounds__(256) void k_normlinear(
    const float* __restrict__ X, const unsigned short* __restrict__ Wb,
    const float* __restrict__ bias, const float* __restrict__ gamma,
    const float* __restrict__ beta, unsigned short* __restrict__ out)
{
  __shared__ unsigned short Xs[128 * 72];
  __shared__ unsigned short Ws[256 * 72];

  const int tid  = threadIdx.x;
  const int lane = tid & 31, wv = tid >> 5;
  const int r15  = lane & 15, hx = lane >> 4;
  const int row0 = blockIdx.x * 128;

  v8f acc[16] = {};

  for (int kc = 0; kc < 4; ++kc) {
    __syncthreads();
    for (int i = tid; i < 2048; i += 256) {
      int n = i >> 3, c = (i & 7) << 3;
      async_ld16(&Ws[n * 72 + c], Wb + (size_t)n * F_ + kc * 64 + c);
    }
    for (int i = tid; i < 128 * 16; i += 256) {
      int y = i >> 4, c4 = (i & 15) << 2;
      float4 v = *(const float4*)(X + (size_t)(row0 + y) * F_ + kc * 64 + c4);
      unsigned short* d = &Xs[y * 72 + c4];
      d[0] = f2bfu(lrelu(v.x)); d[1] = f2bfu(lrelu(v.y));
      d[2] = f2bfu(lrelu(v.z)); d[3] = f2bfu(lrelu(v.w));
    }
    if (kc < 3)
      __builtin_prefetch(X + (size_t)(row0 + (tid >> 1)) * F_ + (kc + 1) * 64 + (tid & 1) * 32, 0, 1);
    async_wait();
    __syncthreads();
    #pragma unroll
    for (int ks = 0; ks < 2; ++ks) {
      const int k0 = ks * 32;
      v16bf a  = frag_a(Xs, wv * 16, 72, k0);
      v16bf b0 = frag_b(Ws,  0, 72, k0);
      v16bf b1 = frag_b(Ws, 16, 72, k0);
      #pragma unroll
      for (int t = 0; t < 16; ++t) {
        acc[t] = wmma_bf16(a, (t & 1) ? b1 : b0, acc[t]);
        int u = t + 2;
        if (u < 16) { if (t & 1) b1 = frag_b(Ws, u * 16, 72, k0); else b0 = frag_b(Ws, u * 16, 72, k0); }
      }
    }
  }

  float s[8] = {}, q[8] = {};
  #pragma unroll
  for (int t = 0; t < 16; ++t) {
    float bn = bias[t * 16 + r15];
    #pragma unroll
    for (int r = 0; r < 8; ++r) {
      float v = acc[t][r] + bn; acc[t][r] = v;
      s[r] += v; q[r] += v * v;
    }
  }
  #pragma unroll
  for (int r = 0; r < 8; ++r) {
    float sv = s[r], qv = q[r];
    #pragma unroll
    for (int m = 1; m < 16; m <<= 1) { sv += __shfl_xor(sv, m, 32); qv += __shfl_xor(qv, m, 32); }
    float mu = sv * (1.f / 256.f);
    s[r] = mu;
    q[r] = rsqrtf(qv * (1.f / 256.f) - mu * mu + 1e-5f);
  }
  #pragma unroll
  for (int t = 0; t < 16; ++t) {
    int n = t * 16 + r15; float g = gamma[n], bb = beta[n];
    #pragma unroll
    for (int r = 0; r < 8; ++r) {
      float v = g * (acc[t][r] - s[r]) * q[r] + bb;
      int row = row0 + wv * 16 + r + 8 * hx;
      out[(size_t)row * 256 + n] = f2bfu(v);
    }
  }
}

// ---------------------------------------------------------------------------
// Kernel 2b: expanded[b,x,y,h] = sum_f xf[b,x,h,f] * yf[b,y,h,f]; K=32 ->
// one WMMA per 16x16 tile, fragments straight from global bf16.
// ---------------------------------------------------------------------------
__global__ __launch_bounds__(256) void k_expand(
    const unsigned short* __restrict__ xf, const unsigned short* __restrict__ yf,
    float* __restrict__ outE)
{
  const int tid  = threadIdx.x;
  const int lane = tid & 31, wv = tid >> 5;
  const int r15  = lane & 15, hx = lane >> 4;
  const int b = blockIdx.x >> 3, h = blockIdx.x & 7;

  const unsigned short* xb = xf + (size_t)b * L_ * HF_;
  const unsigned short* yb = yf + (size_t)b * R_ * HF_;

  v16bf a  = frag_a(xb, wv * 16, HF_, h * 32);
  v16bf b0 = frag_b(yb,  0, HF_, h * 32);
  v16bf b1 = frag_b(yb, 16, HF_, h * 32);
  for (int nt = 0; nt < 32; nt += 2) {
    v8f c0 = {};
    c0 = wmma_bf16(a, b0, c0);
    if (nt + 2 < 32) b0 = frag_b(yb, (nt + 2) * 16, HF_, h * 32);
    #pragma unroll
    for (int r = 0; r < 8; ++r) {
      int x = wv * 16 + r + 8 * hx;
      int y = nt * 16 + r15;
      outE[(((size_t)b * L_ + x) * R_ + y) * H_ + h] = c0[r];
    }
    v8f c1 = {};
    c1 = wmma_bf16(a, b1, c1);
    if (nt + 3 < 32) b1 = frag_b(yb, (nt + 3) * 16, HF_, h * 32);
    #pragma unroll
    for (int r = 0; r < 8; ++r) {
      int x = wv * 16 + r + 8 * hx;
      int y = (nt + 1) * 16 + r15;
      outE[(((size_t)b * L_ + x) * R_ + y) * H_ + h] = c1[r];
    }
  }
}

// ---------------------------------------------------------------------------
// Kernel 3: edge gather -> (16384x512)x(512x64) WMMA GEMM -> bias -> lrelu ->
// scatter-add into llf with global f32 atomics. 128 edges per WG.
// ---------------------------------------------------------------------------
__global__ __launch_bounds__(256) void k_edges(
    const float* __restrict__ lig, const int* __restrict__ eb,
    const int* __restrict__ es, const int* __restrict__ ed,
    const unsigned short* __restrict__ Web,   // bf16 (64,512)
    const float* __restrict__ be, float* __restrict__ outL)
{
  __shared__ unsigned short Es[128 * 136];   // gathered edge features (KC=128)
  __shared__ unsigned short WeS[64 * 136];
  __shared__ int ebS[128], esS[128], edS[128];

  const int tid  = threadIdx.x;
  const int lane = tid & 31, wv = tid >> 5;
  const int r15  = lane & 15, hx = lane >> 4;
  const int e0 = blockIdx.x * 128;

  if (tid < 128) { ebS[tid] = eb[e0 + tid]; esS[tid] = es[e0 + tid]; edS[tid] = ed[e0 + tid]; }

  v8f acc[4] = {};

  for (int kc = 0; kc < 4; ++kc) {
    __syncthreads();
    for (int i = tid; i < 1024; i += 256) {            // 64 rows x 128 cols bf16
      int n = i >> 4, c = (i & 15) << 3;
      async_ld16(&WeS[n * 136 + c], Web + (size_t)n * (2 * F_) + kc * 128 + c);
    }
    // ef columns [kc*128, +128): kc<2 -> src half, kc>=2 -> dst half
    for (int i = tid; i < 128 * 32; i += 256) {
      int e = i >> 5, c4 = (i & 31) << 2;
      int node = (kc < 2) ? esS[e] : edS[e];
      float4 v = *(const float4*)(lig + ((size_t)ebS[e] * L_ + node) * F_ + (kc & 1) * 128 + c4);
      unsigned short* d = &Es[e * 136 + c4];
      d[0] = f2bfu(v.x); d[1] = f2bfu(v.y); d[2] = f2bfu(v.z); d[3] = f2bfu(v.w);
    }
    async_wait();
    __syncthreads();
    #pragma unroll
    for (int ks = 0; ks < 4; ++ks) {
      const int k0 = ks * 32;
      v16bf a  = frag_a(Es, wv * 16, 136, k0);
      v16bf b0 = frag_b(WeS,  0, 136, k0);
      v16bf b1 = frag_b(WeS, 16, 136, k0);
      #pragma unroll
      for (int t = 0; t < 4; ++t) {
        acc[t] = wmma_bf16(a, (t & 1) ? b1 : b0, acc[t]);
        int u = t + 2;
        if (u < 4) { if (t & 1) b1 = frag_b(WeS, u * 16, 136, k0); else b0 = frag_b(WeS, u * 16, 136, k0); }
      }
    }
  }

  #pragma unroll
  for (int t = 0; t < 4; ++t) {
    int n = t * 16 + r15;
    float bb = be[n];
    #pragma unroll
    for (int r = 0; r < 8; ++r) {
      int el = wv * 16 + r + 8 * hx;
      float v = lrelu(acc[t][r] + bb);
      size_t off = (((size_t)ebS[el] * L_ + esS[el]) * L_ + edS[el]) * FE_ + n;
      unsafeAtomicAdd(&outL[off], v);   // global_atomic_add_f32
    }
  }
}

// ---------------------------------------------------------------------------
extern "C" void kernel_launch(void* const* d_in, const int* in_sizes, int n_in,
                              void* d_out, int out_size, void* d_ws, size_t ws_size,
                              hipStream_t stream)
{
  (void)in_sizes; (void)n_in; (void)out_size; (void)ws_size;

  const float*          ll   = (const float*)d_in[0];
  const float*          lig  = (const float*)d_in[1];
  const float*          rec  = (const float*)d_in[2];
  const unsigned char*  mask = (const unsigned char*)d_in[3];
  const int* eb = (const int*)d_in[4];
  const int* es = (const int*)d_in[5];
  const int* ed = (const int*)d_in[6];
  const float* Wc  = (const float*)d_in[7];
  const float* bc  = (const float*)d_in[8];
  const float* gc  = (const float*)d_in[9];
  const float* bec = (const float*)d_in[10];
  const float* Wca = (const float*)d_in[11];
  const float* bca = (const float*)d_in[12];
  const float* Wex = (const float*)d_in[13];
  const float* bex = (const float*)d_in[14];
  const float* gex = (const float*)d_in[15];
  const float* beex = (const float*)d_in[16];
  const float* Wey = (const float*)d_in[17];
  const float* bey = (const float*)d_in[18];
  const float* gey = (const float*)d_in[19];
  const float* beey = (const float*)d_in[20];
  const float* We  = (const float*)d_in[21];
  const float* be  = (const float*)d_in[22];

  float* outC = (float*)d_out;                               // (B,L,HF)
  float* outE = outC + (size_t)B_ * L_ * HF_;                // (B,L,R,H)
  float* outL = outE + (size_t)B_ * L_ * R_ * H_;            // (B,L,L,FE)

  // workspace layout (bf16 elements)
  unsigned short* ws = (unsigned short*)d_ws;
  unsigned short* xf   = ws;                                  // 2048*256
  unsigned short* yf   = xf + (size_t)B_ * L_ * HF_;          // 8192*256
  unsigned short* Wcb  = yf + (size_t)B_ * R_ * HF_;          // 65536
  unsigned short* Wcab = Wcb + 65536;                         // 16*256 (rows 8..15 zero)
  unsigned short* Wexb = Wcab + 4096;                         // 65536
  unsigned short* Weyb = Wexb + 65536;                        // 65536
  unsigned short* Web  = Weyb + 65536;                        // 64*512

  hipMemsetAsync(outL, 0, (size_t)B_ * L_ * L_ * FE_ * sizeof(float), stream);
  hipMemsetAsync(Wcab + 8 * F_, 0, (size_t)8 * F_ * sizeof(unsigned short), stream);

  // one-time weight conversions
  k_cvt_bf16<<<64, 256, 0, stream>>>(Wc,  Wcb,  65536);
  k_cvt_bf16<<<2,  256, 0, stream>>>(Wca, Wcab, 2048);
  k_cvt_bf16<<<64, 256, 0, stream>>>(Wex, Wexb, 65536);
  k_cvt_bf16<<<64, 256, 0, stream>>>(Wey, Weyb, 65536);
  k_cvt_bf16<<<32, 256, 0, stream>>>(We,  Web,  32768);

  k_contract  <<<B_ * L_,         256, 0, stream>>>(ll, mask, Wcb, Wcab, bc, gc, bec, bca, outC);
  k_normlinear<<<(B_ * L_) / 128, 256, 0, stream>>>(lig, Wexb, bex, gex, beex, xf);
  k_normlinear<<<(B_ * R_) / 128, 256, 0, stream>>>(rec, Weyb, bey, gey, beey, yf);
  k_expand    <<<B_ * H_,         256, 0, stream>>>(xf, yf, outE);
  k_edges     <<<E_ / 128,        256, 0, stream>>>(lig, eb, es, ed, Web, be, outL);
}
